// Attention_19559281066066
// MI455X (gfx1250) — compile-verified, matrix operand order checked by reading, not run on hardware
//
#include <hip/hip_runtime.h>
#include <hip/hip_bf16.h>
#include <math.h>

#define DIM 128

typedef __attribute__((ext_vector_type(16))) __bf16 v16bf;
typedef __attribute__((ext_vector_type(8)))  float  v8f;

// Hardware tanh: CDNA5 has V_TANH_F32 (TRANS32). Use the builtin when clang
// declares it, else emit the instruction directly.
static __device__ __forceinline__ float hw_tanhf(float x) {
#if __has_builtin(__builtin_amdgcn_tanhf)
    return __builtin_amdgcn_tanhf(x);
#else
    float r;
    asm volatile("v_tanh_f32 %0, %1" : "=v"(r) : "v"(x));
    return r;
#endif
}

// ---------------------------------------------------------------------------
// Kernel 0: zero the output [B,128] and the per-segment exp-sum accumulator.
// ---------------------------------------------------------------------------
__global__ void zero_kernel(float* __restrict__ out, float* __restrict__ segsum,
                            int nout, int nseg) {
    int i = blockIdx.x * blockDim.x + threadIdx.x;
    if (i < nout) out[i] = 0.0f;
    if (i < nseg) segsum[i] = 0.0f;
}

// ---------------------------------------------------------------------------
// Kernel 1: pack W and U (f32 row-major [K=128, N=128]) into bf16 WMMA
// B-fragment lane order: [mat(2)][kt(4)][nt(8)][lane(32)][16 x bf16].
// 16-bit B layout per 16x16x32 tile (mirrors A layout, col = lane%16):
//   lanes 0-15 : k = kt*32 + {0..7, 16..23}
//   lanes 16-31: k = kt*32 + {8..15, 24..31}
// ---------------------------------------------------------------------------
__global__ void pack_wu_kernel(const float* __restrict__ W,
                               const float* __restrict__ U,
                               __bf16* __restrict__ packed) {
    int t = blockIdx.x * blockDim.x + threadIdx.x;   // 0..2047
    if (t >= 2048) return;
    int lane = t & 31;
    int nt   = (t >> 5) & 7;
    int kt   = (t >> 8) & 3;
    int mat  = (t >> 10) & 1;
    const float* M = mat ? U : W;
    int n    = nt * 16 + (lane & 15);
    int half = lane >> 4;
    int kb1  = kt * 32 + half * 8;
    int kb2  = kt * 32 + 16 + half * 8;
    __bf16* dst = packed + (size_t)t * 16;
#pragma unroll
    for (int j = 0; j < 8; ++j) dst[j]     = (__bf16)M[(size_t)(kb1 + j) * DIM + n];
#pragma unroll
    for (int j = 0; j < 8; ++j) dst[8 + j] = (__bf16)M[(size_t)(kb2 + j) * DIM + n];
}

// ---------------------------------------------------------------------------
// Kernel 2: beta[n] = v . tanh(Q@W + K@U);  e = exp(beta);
//           exp_beta[n] = e;  segsum[bidx[n]] += e.
// 8 waves/block, each wave owns 16 rows; 64 WMMAs per wave.
// ---------------------------------------------------------------------------
__global__ __launch_bounds__(256) void phase1_kernel(
    const float* __restrict__ Q, const float* __restrict__ K,
    const float* __restrict__ vvec, const int* __restrict__ bidx,
    const __bf16* __restrict__ packedB,
    float* __restrict__ exp_beta, float* __restrict__ segsum, int N) {

    __shared__ __bf16 lds_b[2 * 4 * 8 * 32 * 16];   // 64 KB fragment cache

    // Cooperative global -> LDS stage of all B fragments (4096 uint4's).
    {
        const uint4* src = (const uint4*)packedB;
        uint4*       dst = (uint4*)lds_b;
        for (int i = threadIdx.x; i < 4096; i += 256) dst[i] = src[i];
    }
    __syncthreads();

    const int wave    = threadIdx.x >> 5;
    const int lane    = threadIdx.x & 31;
    const int m       = lane & 15;
    const int half    = lane >> 4;
    const int rowbase = (blockIdx.x * 8 + wave) * 16;
    int row  = rowbase + m;
    int rowc = row < N ? row : (N - 1);              // clamp for loads only

    // Build A fragments for Q and K rows (f32 -> bf16, ISA 16-bit A layout).
    v16bf aq[4], ak[4];
#pragma unroll
    for (int kt = 0; kt < 4; ++kt) {
        int b1 = kt * 32 + half * 8;
        int b2 = kt * 32 + 16 + half * 8;
        const float4* q1 = (const float4*)(Q + (size_t)rowc * DIM + b1);
        const float4* q2 = (const float4*)(Q + (size_t)rowc * DIM + b2);
        const float4* k1 = (const float4*)(K + (size_t)rowc * DIM + b1);
        const float4* k2 = (const float4*)(K + (size_t)rowc * DIM + b2);
        float4 qa = q1[0], qb = q1[1], qc = q2[0], qd = q2[1];
        float4 ka = k1[0], kb = k1[1], kc = k2[0], kd = k2[1];
        v16bf tq, tk;
        tq[0]=(__bf16)qa.x; tq[1]=(__bf16)qa.y; tq[2]=(__bf16)qa.z; tq[3]=(__bf16)qa.w;
        tq[4]=(__bf16)qb.x; tq[5]=(__bf16)qb.y; tq[6]=(__bf16)qb.z; tq[7]=(__bf16)qb.w;
        tq[8]=(__bf16)qc.x; tq[9]=(__bf16)qc.y; tq[10]=(__bf16)qc.z; tq[11]=(__bf16)qc.w;
        tq[12]=(__bf16)qd.x; tq[13]=(__bf16)qd.y; tq[14]=(__bf16)qd.z; tq[15]=(__bf16)qd.w;
        tk[0]=(__bf16)ka.x; tk[1]=(__bf16)ka.y; tk[2]=(__bf16)ka.z; tk[3]=(__bf16)ka.w;
        tk[4]=(__bf16)kb.x; tk[5]=(__bf16)kb.y; tk[6]=(__bf16)kb.z; tk[7]=(__bf16)kb.w;
        tk[8]=(__bf16)kc.x; tk[9]=(__bf16)kc.y; tk[10]=(__bf16)kc.z; tk[11]=(__bf16)kc.w;
        tk[12]=(__bf16)kd.x; tk[13]=(__bf16)kd.y; tk[14]=(__bf16)kd.z; tk[15]=(__bf16)kd.w;
        aq[kt] = tq;
        ak[kt] = tk;
    }

    // Preload this lane's slice of v: column nt*16 + m for each ntile.
    float vreg[8];
#pragma unroll
    for (int nt = 0; nt < 8; ++nt) vreg[nt] = vvec[nt * 16 + m];

    const v16bf* frag = (const v16bf*)lds_b;   // [mat][kt][nt][lane]
    float acc[8] = {0.f, 0.f, 0.f, 0.f, 0.f, 0.f, 0.f, 0.f};

#pragma unroll
    for (int nt = 0; nt < 8; ++nt) {
        v8f c = {0.f, 0.f, 0.f, 0.f, 0.f, 0.f, 0.f, 0.f};
#pragma unroll
        for (int kt = 0; kt < 4; ++kt) {
            v16bf bw = frag[((0 * 4 + kt) * 8 + nt) * 32 + lane];
            c = __builtin_amdgcn_wmma_f32_16x16x32_bf16(
                    false, aq[kt], false, bw, (short)0, c, false, false);
        }
#pragma unroll
        for (int kt = 0; kt < 4; ++kt) {
            v16bf bu = frag[((1 * 4 + kt) * 8 + nt) * 32 + lane];
            c = __builtin_amdgcn_wmma_f32_16x16x32_bf16(
                    false, ak[kt], false, bu, (short)0, c, false, false);
        }
        // C layout: lane column = nt*16 + m ; VGPR g = row (g + half*8).
        float vv = vreg[nt];
#pragma unroll
        for (int g = 0; g < 8; ++g) acc[g] += hw_tanhf(c[g]) * vv;
    }

    // Reduce over the 16 columns held by each 16-lane half (xor masks 1..8
    // never cross the half boundary at bit 4).
#pragma unroll
    for (int g = 0; g < 8; ++g) {
        float s = acc[g];
        for (int off = 8; off > 0; off >>= 1) s += __shfl_xor(s, off, 32);
        acc[g] = s;
    }

    if (m == 0) {  // lanes 0 and 16: rows rowbase+half*8+{0..7}
#pragma unroll
        for (int g = 0; g < 8; ++g) {
            int r = rowbase + half * 8 + g;
            if (r < N) {
                float e = __expf(acc[g]);      // raw exp, faithful to reference
                exp_beta[r] = e;
                atomicAdd(&segsum[bidx[r]], e);
            }
        }
    }
}

// ---------------------------------------------------------------------------
// Kernel 3: out[bidx[n], :] += (exp_beta[n] / segsum[bidx[n]]) * V[n, :]
// batch_index is sorted -> accumulate runs in registers, atomics only at
// segment boundaries. One wave per 128-row chunk; lane owns 4 columns.
// ---------------------------------------------------------------------------
__global__ __launch_bounds__(256) void phase2_kernel(
    const float* __restrict__ V, const int* __restrict__ bidx,
    const float* __restrict__ exp_beta, const float* __restrict__ segsum,
    float* __restrict__ out, int N) {

    const int ROWS = 128;
    int wave  = threadIdx.x >> 5;
    int lane  = threadIdx.x & 31;
    int start = (blockIdx.x * 8 + wave) * ROWS;
    if (start >= N) return;
    int end = start + ROWS;
    if (end > N) end = N;
    int col = lane * 4;

    float ax = 0.f, ay = 0.f, az = 0.f, aw = 0.f;
    int   cur    = bidx[start];
    float curInv = 1.0f / segsum[cur];

    for (int r = start; r < end; ++r) {
        int s = bidx[r];
        if (s != cur) {
            float* o = out + (size_t)cur * DIM + col;
            atomicAdd(o + 0, ax); atomicAdd(o + 1, ay);
            atomicAdd(o + 2, az); atomicAdd(o + 3, aw);
            ax = ay = az = aw = 0.f;
            cur = s;
            curInv = 1.0f / segsum[s];
        }
        float a = exp_beta[r] * curInv;
        float4 vr = *(const float4*)(V + (size_t)r * DIM + col);
        ax += a * vr.x; ay += a * vr.y; az += a * vr.z; aw += a * vr.w;
    }
    float* o = out + (size_t)cur * DIM + col;
    atomicAdd(o + 0, ax); atomicAdd(o + 1, ay);
    atomicAdd(o + 2, az); atomicAdd(o + 3, aw);
}

// ---------------------------------------------------------------------------
// Host-side launcher.
// Inputs (setup_inputs order): Q[N,128], K[N,128], V[N,128], batch_index[N],
//                              W[128,128], U[128,128], v[128]
// Output: [512,128] f32.  Workspace layout:
//   [0, 4N)            exp_beta  (f32[N])
//   [4N, 4N+4B)        segsum    (f32[B])
//   [4N+4B, +64KB)     packed bf16 B fragments for W and U
// ---------------------------------------------------------------------------
extern "C" void kernel_launch(void* const* d_in, const int* in_sizes, int n_in,
                              void* d_out, int out_size, void* d_ws, size_t ws_size,
                              hipStream_t stream) {
    const float* Q    = (const float*)d_in[0];
    const float* K    = (const float*)d_in[1];
    const float* V    = (const float*)d_in[2];
    const int*   bidx = (const int*)  d_in[3];
    const float* W    = (const float*)d_in[4];
    const float* U    = (const float*)d_in[5];
    const float* v    = (const float*)d_in[6];

    const int N = in_sizes[3];          // 500000
    const int B = out_size / DIM;       // 512
    float* out = (float*)d_out;

    char*   ws       = (char*)d_ws;
    float*  exp_beta = (float*)ws;
    float*  segsum   = (float*)(ws + (size_t)N * 4);
    __bf16* packed   = (__bf16*)(ws + (size_t)N * 4 + (size_t)B * 4);

    zero_kernel<<<(out_size + 255) / 256, 256, 0, stream>>>(out, segsum, out_size, B);
    pack_wu_kernel<<<8, 256, 0, stream>>>(W, U, packed);

    int nblk1 = (N + 127) / 128;        // 8 waves x 16 rows per block
    phase1_kernel<<<nblk1, 256, 0, stream>>>(Q, K, v, bidx, packed,
                                             exp_beta, segsum, N);

    int nblk2 = (N + 8 * 128 - 1) / (8 * 128);
    phase2_kernel<<<nblk2, 256, 0, stream>>>(V, bidx, exp_beta, segsum, out, N);
}
